// SparseConv3d_31628139167766
// MI455X (gfx1250) — matrix operand order repaired and checked by reading, not compiled, not run on hardware
//
#include <hip/hip_runtime.h>
#include <hip/hip_bf16.h>

// Implicit-GEMM sparse-masked 3D conv for MI455X (gfx1250, wave32).
// bf16x3 split (hi+lo) on v_wmma_f32_16x16x32_bf16 -> ~fp32 accuracy at
// tensor-core rates. One 512-thread block per (batch, z, y) x-row computes
// the full 64(cout) x 64(vox) output tile; 16 waves each own one 16x16 tile.
// Weights are pre-split to bf16 hi/lo planes in d_ws (padded to the LDS tile
// layout) by a prep kernel; the main kernel stages them via the Tensor Data
// Mover (tensor_load_to_lds + s_wait_tensorcnt).

typedef __attribute__((ext_vector_type(16))) __bf16 v16bf;
typedef __attribute__((ext_vector_type(8)))  float  v8f;
typedef __attribute__((ext_vector_type(4)))  unsigned int v4u;
typedef __attribute__((ext_vector_type(8)))  int v8i;
typedef __attribute__((ext_vector_type(4)))  int v4i;

#define RDIM 64
#define CDIM 64
#define R2   (RDIM * RDIM)          // 4096
#define R3   (RDIM * RDIM * RDIM)   // 262144
#define LDA  72                     // padded LDS row stride in halves (144 B = 9*16 B)
#define NTHREADS 512
#define PLANE_HALF (CDIM * LDA)     // 4608 halves = 9216 B per tap per plane
#define PLANE_BYTES (PLANE_HALF * 2)
#define NTAPS 27

union FragU { uint4 q[2]; v16bf v; };

// ---- TDM: 1-D bulk copy global -> LDS via Tensor DMA descriptor ------------
// Group0: count=1, lds_addr, global_addr[56:0], type=2.
// Group1: data_size=4B, tensor_dim0 = tile_dim0 = nbytes/4 dwords, dim1=1.
__device__ inline void tdm_copy_to_lds(const void* gsrc, unsigned lds_off,
                                       unsigned nbytes) {
    const unsigned ndw = nbytes >> 2;                    // dwords (fits 16 bits)
    const unsigned long long ga = (unsigned long long)(uintptr_t)gsrc;
    v4u g0;
    g0[0] = 1u;                                          // count=1, user mode
    g0[1] = lds_off;                                     // lds_addr (bytes)
    g0[2] = (unsigned)ga;                                // global_addr[31:0]
    g0[3] = (unsigned)(((ga >> 32) & 0x01ffffffull)      // global_addr[56:32]
                       | (2u << 30));                    // type = 2 ("image")
    v8i g1;
    g1[0] = (int)(2u << 16);                 // wg_mask=0, data_size=2 (4 B)
    g1[1] = (int)((ndw & 0xffffu) << 16);    // tensor_dim0[15:0]  @ bits 63:48
    g1[2] = (int)((ndw >> 16) | (1u << 16)); // tensor_dim0[31:16], tensor_dim1=1
    g1[3] = (int)((ndw & 0xffffu) << 16);    // tile_dim0 = ndw    @ bits 127:112
    g1[4] = 1;                               // tile_dim1 = 1
    g1[5] = (int)ndw;                        // tensor_dim0_stride[31:0]
    g1[6] = 0;
    g1[7] = 0;
    v4i z4 = {0, 0, 0, 0};
#if defined(__clang_major__) && (__clang_major__ >= 23)
    v8i z8 = {0, 0, 0, 0, 0, 0, 0, 0};
    __builtin_amdgcn_tensor_load_to_lds(g0, g1, z4, z4, z8, 0);
#else
    __builtin_amdgcn_tensor_load_to_lds(g0, g1, z4, z4, 0);
#endif
}

// ---- prep: split weight f32 -> bf16 hi/lo, padded to the LDS tile layout ---
// ws layout: hi plane [tap][cout][LDA], then lo plane, each NTAPS*PLANE_HALF.
__global__ __launch_bounds__(256)
void prep_weights_bf16(const float* __restrict__ w,
                       unsigned short* __restrict__ wsHiRaw,
                       unsigned short* __restrict__ wsLoRaw) {
    int idx = blockIdx.x * 256 + threadIdx.x;
    if (idx >= NTAPS * CDIM * CDIM) return;
    int t  = idx / (CDIM * CDIM);
    int r  = idx - t * (CDIM * CDIM);
    int co = r >> 6, ci = r & 63;
    float f = w[(size_t)(co * CDIM + ci) * NTAPS + t];   // tap t = kz*9+ky*3+kx
    __bf16 h = (__bf16)f;
    __bf16 l = (__bf16)(f - (float)h);
    ((__bf16*)wsHiRaw)[(size_t)t * PLANE_HALF + co * LDA + ci] = h;
    ((__bf16*)wsLoRaw)[(size_t)t * PLANE_HALF + co * LDA + ci] = l;
}

__global__ __launch_bounds__(NTHREADS)
void sparse_conv3d_igemm_bf16x3(const float* __restrict__ x,
                                const float* __restrict__ w,
                                const unsigned short* __restrict__ wsHiRaw,
                                const unsigned short* __restrict__ wsLoRaw,
                                float* __restrict__ out,
                                int useTdm)
{
    __shared__ __align__(16) __bf16 sAhi[PLANE_HALF];
    __shared__ __align__(16) __bf16 sAlo[PLANE_HALF];
    __shared__ __align__(16) __bf16 sBhi[68 * LDA];
    __shared__ __align__(16) __bf16 sBlo[68 * LDA];
    __shared__ float smask[RDIM];

    const int tid = threadIdx.x;
    const int bid = blockIdx.x;
    const int b  = bid >> 12;        // 4096 (z,y) rows per batch
    const int zy = bid & 4095;
    const int z  = zy >> 6;
    const int y  = zy & 63;

    const float* xb = x + (size_t)b * CDIM * R3;

    // Per-voxel mask: sum_c |x[b,c,z,y,v]| != 0  (matches reference).
    if (tid < RDIM) {
        float s = 0.f;
        const float* p = xb + (size_t)z * R2 + (size_t)y * RDIM + tid;
        #pragma unroll 4
        for (int c = 0; c < CDIM; ++c) s += fabsf(p[(size_t)c * R3]);
        smask[tid] = (s != 0.f) ? 1.f : 0.f;
    }

    const int wave = tid >> 5;
    const int lane = tid & 31;
    const int half = lane >> 4;      // 0: lanes 0-15, 1: lanes 16-31
    const int l16  = lane & 15;
    const int tm   = wave >> 2;      // cout tile 0..3
    const int tn   = wave & 3;       // vox  tile 0..3

    const unsigned ldsAhiOff = (unsigned)(uintptr_t)&sAhi[0];
    const unsigned ldsAloOff = (unsigned)(uintptr_t)&sAlo[0];

    v8f acc = {};

    for (int t9 = 0; t9 < 9; ++t9) {
        const int kz = t9 / 3, ky = t9 % 3;
        const int zz = z + kz - 1;
        const int yy = y + ky - 1;
        const bool planeOK = ((unsigned)zz < RDIM) && ((unsigned)yy < RDIM);

        __syncthreads();   // previous iteration's WMMAs done before restaging B

        // Stage B: rows v = 0..65 hold x' = v-1 (SAME padding -> zeros OOB),
        // stored [v][cin] so a dx tap shift is a row shift (keeps 16B align).
        for (int p = 0; p < 9; ++p) {
            int idx = tid + p * NTHREADS;
            if (idx < 66 * CDIM) {
                int v  = idx % 66;       // consecutive lanes -> consecutive x: coalesced
                int c  = idx / 66;
                int xp = v - 1;
                float f = 0.f;
                if (planeOK && (unsigned)xp < RDIM)
                    f = xb[(size_t)c * R3 + (size_t)zz * R2 + (size_t)yy * RDIM + xp];
                __bf16 h = (__bf16)f;
                __bf16 l = (__bf16)(f - (float)h);
                sBhi[v * LDA + c] = h;
                sBlo[v * LDA + c] = l;
            }
        }

        for (int kx = 0; kx < 3; ++kx) {
            __syncthreads();   // prior compute done before overwriting A
            if (useTdm) {
                // A tiles pre-split/pre-padded in d_ws: one TDM DMA per plane.
                if (wave == 0) {
                    const int t = t9 * 3 + kx;   // kz*9+ky*3+kx
                    tdm_copy_to_lds((const char*)wsHiRaw + (size_t)t * PLANE_BYTES,
                                    ldsAhiOff, PLANE_BYTES);
                    tdm_copy_to_lds((const char*)wsLoRaw + (size_t)t * PLANE_BYTES,
                                    ldsAloOff, PLANE_BYTES);
                    __builtin_amdgcn_s_wait_tensorcnt(0);
                }
            } else {
                // Fallback: stage + convert weights in-kernel.
                for (int p = 0; p < 8; ++p) {
                    int idx = tid + p * NTHREADS;    // exactly 4096 elements
                    int ci = idx & 63;
                    int co = idx >> 6;
                    float f = w[((co * CDIM + ci) * 9 + (kz * 3 + ky)) * 3 + kx];
                    __bf16 h = (__bf16)f;
                    __bf16 l = (__bf16)(f - (float)h);
                    sAhi[co * LDA + ci] = h;
                    sAlo[co * LDA + ci] = l;
                }
            }
            __syncthreads();   // staging visible to all waves

            #pragma unroll
            for (int ck = 0; ck < 2; ++ck) {     // cin 0..31, 32..63
                // A fragment (16x32, MxK): lane<16 holds K 0..15,
                // lane>=16 holds K 16..31 of its row M = tm*16+l16.
                const int  arow = tm * 16 + l16;
                const int  koff = ck * 32 + half * 16;
                const __bf16* pah = &sAhi[arow * LDA + koff];
                const __bf16* pal = &sAlo[arow * LDA + koff];
                FragU ahi, alo, bhi, blo;
                ahi.q[0] = *(const uint4*)(pah);
                ahi.q[1] = *(const uint4*)(pah + 8);    // +8 halves = K+8..15
                alo.q[0] = *(const uint4*)(pal);
                alo.q[1] = *(const uint4*)(pal + 8);

                // B fragment (32x16, KxN): column N = tn*16+l16; lane<16 holds
                // K 0..15, lane>=16 holds K 16..31 (contiguous in staged row).
                const int  brow = tn * 16 + l16 + kx;   // x' = vox + kx - 1 -> row x'+1
                const int  boff = ck * 32 + half * 16;
                const __bf16* pbh = &sBhi[brow * LDA + boff];
                const __bf16* pbl = &sBlo[brow * LDA + boff];
                bhi.q[0] = *(const uint4*)(pbh);
                bhi.q[1] = *(const uint4*)(pbh + 8);
                blo.q[0] = *(const uint4*)(pbl);
                blo.q[1] = *(const uint4*)(pbl + 8);

                // fp32 ~= hi*hi + hi*lo + lo*hi (lo*lo term negligible, 2^-16)
                acc = __builtin_amdgcn_wmma_f32_16x16x32_bf16(
                          false, ahi.v, false, bhi.v, (short)0, acc, false, false);
                acc = __builtin_amdgcn_wmma_f32_16x16x32_bf16(
                          false, alo.v, false, bhi.v, (short)0, acc, false, false);
                acc = __builtin_amdgcn_wmma_f32_16x16x32_bf16(
                          false, ahi.v, false, blo.v, (short)0, acc, false, false);
            }
        }
    }

    __syncthreads();
    // Epilogue: C/D layout -> lane l, reg r: cout = tm*16 + 8*(l/16) + r,
    // vox = tn*16 + l%16. Apply mask, store fp32.
    const int   vox = tn * 16 + l16;
    const float m   = smask[vox];
    #pragma unroll
    for (int r = 0; r < 8; ++r) {
        int cout = tm * 16 + half * 8 + r;
        out[((size_t)(b * CDIM + cout)) * R3 + (size_t)z * R2 + (size_t)y * RDIM + vox]
            = acc[r] * m;
    }
}

extern "C" void kernel_launch(void* const* d_in, const int* in_sizes, int n_in,
                              void* d_out, int out_size, void* d_ws, size_t ws_size,
                              hipStream_t stream) {
    (void)in_sizes; (void)n_in; (void)out_size;
    const float* x = (const float*)d_in[0];   // [4,64,64,64,64] f32
    const float* w = (const float*)d_in[1];   // [64,64,3,3,3]  f32
    float* out = (float*)d_out;               // [4,64,64,64,64] f32

    const size_t planeBytes = (size_t)NTAPS * PLANE_BYTES;     // 248,832 B
    const size_t wsNeed     = 2 * planeBytes;                   // ~486 KB
    int useTdm = (d_ws != nullptr && ws_size >= wsNeed) ? 1 : 0;

    unsigned short* wsHi = (unsigned short*)d_ws;
    unsigned short* wsLo = (unsigned short*)((char*)d_ws + planeBytes);

    if (useTdm) {
        const int n = NTAPS * CDIM * CDIM;                      // 110,592
        prep_weights_bf16<<<(n + 255) / 256, 256, 0, stream>>>(w, wsHi, wsLo);
    }

    dim3 grid(4 * RDIM * RDIM);   // one block per (b, z, y) x-row
    dim3 block(NTHREADS);         // 16 wave32
    sparse_conv3d_igemm_bf16x3<<<grid, block, 0, stream>>>(x, w, wsHi, wsLo, out, useTdm);
}
